// RNN_23149873725703
// MI455X (gfx1250) — matrix operand order repaired and checked by reading, not compiled, not run on hardware
//
#include <hip/hip_runtime.h>
#include <math.h>

#define B_  64
#define S_  512
#define IN_ 1024
#define H_  1024

typedef __bf16 bf16;
typedef bf16  v16bf __attribute__((ext_vector_type(16)));
typedef bf16  v8bf  __attribute__((ext_vector_type(8)));
typedef float v8f   __attribute__((ext_vector_type(8)));

// ---------------------------------------------------------------------------
// WMMA fragment loaders, per CDNA5 ISA 7.12.2 layouts (wave32).
// A: 16x32 bf16, row m = lane&15; lanes 0-15 hold K {0..7,16..23},
//    lanes 16-31 hold K {8..15,24..31}  -> two contiguous 16B loads.
// B: 32x16 bf16 for C = A * W^T, B[k,n] = W[nbase+n, k]; lane n = lane&15,
//    same K split as A: lanes 0-15 K {0..7,16..23}, lanes 16-31 K {8..15,24..31} -> 2x16B loads.
// ---------------------------------------------------------------------------
__device__ __forceinline__ v16bf load_a_frag(const bf16* __restrict__ A, int lda,
                                             int mbase, int k0, int lane) {
  const int m  = mbase + (lane & 15);
  const int kb = (lane >> 4) * 8;
  const bf16* p = A + (size_t)m * lda + (k0 + kb);
  v8bf lo = *(const v8bf*)(p);
  v8bf hi = *(const v8bf*)(p + 16);
  v16bf a;
#pragma unroll
  for (int i = 0; i < 8; ++i) { a[i] = lo[i]; a[i + 8] = hi[i]; }
  return a;
}

__device__ __forceinline__ v16bf load_b_frag(const bf16* __restrict__ W, int ldw,
                                             int nbase, int k0, int lane) {
  const int n  = nbase + (lane & 15);
  const int kb = (lane >> 4) * 8;
  const bf16* p = W + (size_t)n * ldw + (k0 + kb);
  v8bf lo = *(const v8bf*)(p);
  v8bf hi = *(const v8bf*)(p + 16);
  v16bf b;
#pragma unroll
  for (int i = 0; i < 8; ++i) { b[i] = lo[i]; b[i + 8] = hi[i]; }
  return b;
}

#define WMMA_BF16(a, b, c) __builtin_amdgcn_wmma_f32_16x16x32_bf16( \
    false, (a), false, (b), (short)0, (c), false, false)

// 2x-unrolled ping-pong pipelined 16x(16*NJ) tile of C += A(16xK) * W^T.
// Two fragment register sets alternate each half-iteration: loads write
// directly into the registers consumed one iteration later -> no v_mov
// copies, VMEM overlaps the matrix pipe with partial s_wait_loadcnt.
// Requires K % 64 == 0.
template<int NJ>
__device__ __forceinline__ void gemm_tile(const bf16* __restrict__ A, int lda,
                                          const bf16* __restrict__ W, int ldw,
                                          int mbase, int nbase, int K, int lane,
                                          v8f* acc) {
  v16bf a0 = load_a_frag(A, lda, mbase, 0, lane);
  v16bf b0[NJ];
#pragma unroll
  for (int j = 0; j < NJ; ++j) b0[j] = load_b_frag(W, ldw, nbase + 16 * j, 0, lane);
  v16bf a1 = load_a_frag(A, lda, mbase, 32, lane);
  v16bf b1[NJ];
#pragma unroll
  for (int j = 0; j < NJ; ++j) b1[j] = load_b_frag(W, ldw, nbase + 16 * j, 32, lane);

  for (int k = 64; k < K; k += 64) {
    __builtin_prefetch(A + (size_t)(mbase + (lane & 15)) * lda + (k + 64), 0, 3);
#pragma unroll
    for (int j = 0; j < NJ; ++j) acc[j] = WMMA_BF16(a0, b0[j], acc[j]);
    a0 = load_a_frag(A, lda, mbase, k, lane);
#pragma unroll
    for (int j = 0; j < NJ; ++j) b0[j] = load_b_frag(W, ldw, nbase + 16 * j, k, lane);
#pragma unroll
    for (int j = 0; j < NJ; ++j) acc[j] = WMMA_BF16(a1, b1[j], acc[j]);
    a1 = load_a_frag(A, lda, mbase, k + 32, lane);
#pragma unroll
    for (int j = 0; j < NJ; ++j) b1[j] = load_b_frag(W, ldw, nbase + 16 * j, k + 32, lane);
  }
#pragma unroll
  for (int j = 0; j < NJ; ++j) acc[j] = WMMA_BF16(a0, b0[j], acc[j]);
#pragma unroll
  for (int j = 0; j < NJ; ++j) acc[j] = WMMA_BF16(a1, b1[j], acc[j]);
}

// ---------------------------------------------------------------------------
// Elementwise helpers
// ---------------------------------------------------------------------------
__global__ __launch_bounds__(256) void cvt_f32_bf16(const float* __restrict__ in,
                                                    bf16* __restrict__ out, size_t n) {
  size_t i = (size_t)blockIdx.x * blockDim.x + threadIdx.x;
  size_t stride = (size_t)gridDim.x * blockDim.x;
  for (; i < n; i += stride) out[i] = (bf16)in[i];
}

__global__ __launch_bounds__(256) void zero_f4(float4* __restrict__ p, size_t n4) {
  size_t i = (size_t)blockIdx.x * blockDim.x + threadIdx.x;
  size_t stride = (size_t)gridDim.x * blockDim.x;
  float4 z = make_float4(0.f, 0.f, 0.f, 0.f);
  for (; i < n4; i += stride) p[i] = z;
}

// ---------------------------------------------------------------------------
// Big time-parallel GEMM: xp0[(b*S+s), n] = sum_k x[(b*S+s), k]*Wxh0[n,k] + bxh0[n]
// One wave per 16x64 tile. 32768 waves total.
// ---------------------------------------------------------------------------
__global__ __launch_bounds__(256) void gemm0_kernel(const bf16* __restrict__ xbf,
                                                    const bf16* __restrict__ Wbf,
                                                    const float* __restrict__ bias,
                                                    float* __restrict__ xp0) {
  const int lane = threadIdx.x & 31;
  const int wg   = blockIdx.x * 8 + (threadIdx.x >> 5);   // [0, 32768)
  const int mbase = (wg >> 4) * 16;                       // row tile over B*S
  const int nbase = (wg & 15) * 64;                       // col chunk
  v8f acc[4] = {};
  gemm_tile<4>(xbf, IN_, Wbf, IN_, mbase, nbase, IN_, lane, acc);
#pragma unroll
  for (int j = 0; j < 4; ++j) {
    const int col = nbase + 16 * j + (lane & 15);
    const float bv = bias[col];
#pragma unroll
    for (int r = 0; r < 8; ++r) {
      const int row = mbase + r + 8 * (lane >> 4);
      xp0[(size_t)row * H_ + col] = acc[j][r] + bv;
    }
  }
}

// ---------------------------------------------------------------------------
// Device-scope grid barrier (persistent kernel). Agent-scope acq/rel atomics
// lower to the proper CDNA5 cache controls; spin uses s_sleep.
// ---------------------------------------------------------------------------
__device__ __forceinline__ void grid_barrier(int* cnt, int* gen, int nblk) {
  __syncthreads();
  if (threadIdx.x == 0) {
    const int g = __hip_atomic_load(gen, __ATOMIC_RELAXED, __HIP_MEMORY_SCOPE_AGENT);
    const int arrived =
        __hip_atomic_fetch_add(cnt, 1, __ATOMIC_ACQ_REL, __HIP_MEMORY_SCOPE_AGENT);
    if (arrived == nblk - 1) {
      __hip_atomic_store(cnt, 0, __ATOMIC_RELAXED, __HIP_MEMORY_SCOPE_AGENT);
      __hip_atomic_fetch_add(gen, 1, __ATOMIC_ACQ_REL, __HIP_MEMORY_SCOPE_AGENT);
    } else {
      while (__hip_atomic_load(gen, __ATOMIC_ACQUIRE, __HIP_MEMORY_SCOPE_AGENT) == g)
        __builtin_amdgcn_s_sleep(2);
    }
  }
  __syncthreads();
}

// ---------------------------------------------------------------------------
// Persistent scan kernel: the whole 512-step recurrence in ONE launch.
// 24 blocks x 256 threads = 192 waves, one grid barrier per time step.
// Pipeline across layers:
//   iter tt: group0 (waves 0..63)   : h0(tt)   = tanh(xp0[:,tt,:] + h0(tt-1)@Whh0^T + bhh0)
//            group1 (waves 64..191) : h1(tt-1) = tanh(h0(tt-1)@Wxh1^T + h1(tt-2)@Whh1^T + b)
// h0/h1 are double-buffered: h0(t) lives in buffer (t&1), h1(t) in buffer (t&1).
// ---------------------------------------------------------------------------
__global__ __launch_bounds__(256) void scan_kernel(
    const float* __restrict__ xp0,
    const bf16* __restrict__ Whh0bf, const float* __restrict__ bhh0,
    const bf16* __restrict__ Wxh1bf, const float* __restrict__ bxh1,
    const bf16* __restrict__ Whh1bf, const float* __restrict__ bhh1,
    bf16* __restrict__ h0b0, bf16* __restrict__ h0b1,
    bf16* __restrict__ h1b0, bf16* __restrict__ h1b1,
    float* __restrict__ h0f32, float* __restrict__ h1f32,
    float* __restrict__ out, int* cnt, int* gen) {
  const int lane = threadIdx.x & 31;
  const int wid  = blockIdx.x * 8 + (threadIdx.x >> 5);   // [0, 192)
  const int nblk = gridDim.x;

  for (int tt = 0; tt <= S_; ++tt) {
    // buffer holding h0(tt-1) / h1(tt-2): index (tt+1)&1 / tt&1; write index tt&1 / (tt+1)&1
    const bf16* h0prev = (tt & 1) ? h0b0 : h0b1;
    if (wid < 64) {
      if (tt < S_) {
        bf16* h0out = (tt & 1) ? h0b1 : h0b0;
        const int mbase = (wid >> 4) * 16;
        const int nbase = (wid & 15) * 64;
        v8f acc[4] = {};
        gemm_tile<4>(h0prev, H_, Whh0bf, H_, mbase, nbase, H_, lane, acc);
#pragma unroll
        for (int j = 0; j < 4; ++j) {
          const int col = nbase + 16 * j + (lane & 15);
          const float bv = bhh0[col];
#pragma unroll
          for (int r = 0; r < 8; ++r) {
            const int b = mbase + r + 8 * (lane >> 4);
            const float h =
                tanhf(acc[j][r] + xp0[((size_t)b * S_ + tt) * H_ + col] + bv);
            h0f32[(size_t)b * H_ + col] = h;
            h0out[(size_t)b * H_ + col] = (bf16)h;
          }
        }
      }
    } else {
      if (tt >= 1) {
        const bf16* h1prev = (tt & 1) ? h1b1 : h1b0;   // h1(tt-2): index tt&1
        bf16*       h1out  = (tt & 1) ? h1b0 : h1b1;   // h1(tt-1): index (tt+1)&1
        const int tile  = wid - 64;                    // [0, 128)
        const int mbase = (tile >> 5) * 16;
        const int nbase = (tile & 31) * 32;
        const int tout  = tt - 1;
        v8f acc[2] = {};
        gemm_tile<2>(h0prev, H_, Wxh1bf, H_, mbase, nbase, H_, lane, acc);
        gemm_tile<2>(h1prev, H_, Whh1bf, H_, mbase, nbase, H_, lane, acc);
#pragma unroll
        for (int j = 0; j < 2; ++j) {
          const int col = nbase + 16 * j + (lane & 15);
          const float bv = bxh1[col] + bhh1[col];
#pragma unroll
          for (int r = 0; r < 8; ++r) {
            const int b = mbase + r + 8 * (lane >> 4);
            const float h = tanhf(acc[j][r] + bv);
            out[((size_t)b * S_ + tout) * H_ + col] = h;
            h1f32[(size_t)b * H_ + col] = h;
            h1out[(size_t)b * H_ + col] = (bf16)h;
          }
        }
      }
    }
    grid_barrier(cnt, gen, nblk);
  }
}

__global__ __launch_bounds__(256) void final_kernel(const float* __restrict__ h0f32,
                                                    const float* __restrict__ h1f32,
                                                    float* __restrict__ dst) {
  const int n = B_ * H_;
  int i = blockIdx.x * blockDim.x + threadIdx.x;
  if (i < n) {
    dst[i]     = h0f32[i];
    dst[n + i] = h1f32[i];
  }
}

// ---------------------------------------------------------------------------
extern "C" void kernel_launch(void* const* d_in, const int* in_sizes, int n_in,
                              void* d_out, int out_size, void* d_ws, size_t ws_size,
                              hipStream_t stream) {
  (void)in_sizes; (void)n_in; (void)out_size; (void)ws_size;
  const float* x    = (const float*)d_in[0];
  const float* Wxh0 = (const float*)d_in[1];
  const float* bxh0 = (const float*)d_in[2];
  const float* Whh0 = (const float*)d_in[3];
  const float* bhh0 = (const float*)d_in[4];
  const float* Wxh1 = (const float*)d_in[5];
  const float* bxh1 = (const float*)d_in[6];
  const float* Whh1 = (const float*)d_in[7];
  const float* bhh1 = (const float*)d_in[8];
  float* out = (float*)d_out;

  // ---- workspace carve (all sizes are multiples of 256B) ----
  char* w = (char*)d_ws;
  size_t off = 0;
  auto carve = [&](size_t bytes) { char* p = w + off; off += (bytes + 255) & ~(size_t)255; return p; };
  bf16*  x_bf    = (bf16*)carve((size_t)B_ * S_ * IN_ * sizeof(bf16));   // 64 MB
  float* xp0     = (float*)carve((size_t)B_ * S_ * H_ * sizeof(float));  // 128 MB
  bf16*  Wxh0_bf = (bf16*)carve((size_t)H_ * IN_ * sizeof(bf16));        // 2 MB each
  bf16*  Whh0_bf = (bf16*)carve((size_t)H_ * H_ * sizeof(bf16));
  bf16*  Wxh1_bf = (bf16*)carve((size_t)H_ * H_ * sizeof(bf16));
  bf16*  Whh1_bf = (bf16*)carve((size_t)H_ * H_ * sizeof(bf16));
  // contiguous zero-initialized state region:
  float* h0f32   = (float*)carve((size_t)B_ * H_ * sizeof(float));
  float* h1f32   = (float*)carve((size_t)B_ * H_ * sizeof(float));
  bf16*  h0b0    = (bf16*)carve((size_t)B_ * H_ * sizeof(bf16));
  bf16*  h0b1    = (bf16*)carve((size_t)B_ * H_ * sizeof(bf16));
  bf16*  h1b0    = (bf16*)carve((size_t)B_ * H_ * sizeof(bf16));
  bf16*  h1b1    = (bf16*)carve((size_t)B_ * H_ * sizeof(bf16));
  int*   sync    = (int*)carve(256);                                     // cnt, gen

  // ---- precision conversion (bf16 weights stay resident in 192MB L2) ----
  cvt_f32_bf16<<<1024, 256, 0, stream>>>(x,    x_bf,    (size_t)B_ * S_ * IN_);
  cvt_f32_bf16<<<256,  256, 0, stream>>>(Wxh0, Wxh0_bf, (size_t)H_ * IN_);
  cvt_f32_bf16<<<256,  256, 0, stream>>>(Whh0, Whh0_bf, (size_t)H_ * H_);
  cvt_f32_bf16<<<256,  256, 0, stream>>>(Wxh1, Wxh1_bf, (size_t)H_ * H_);
  cvt_f32_bf16<<<256,  256, 0, stream>>>(Whh1, Whh1_bf, (size_t)H_ * H_);

  // zero states + barrier vars (contiguous region carved above)
  {
    size_t state_bytes = 2 * (size_t)B_ * H_ * sizeof(float)
                       + 4 * (size_t)B_ * H_ * sizeof(bf16) + 256;
    zero_f4<<<288, 256, 0, stream>>>((float4*)h0f32, state_bytes / 16);
  }

  // ---- time-parallel input projection: 69 GFLOP WMMA GEMM ----
  gemm0_kernel<<<4096, 256, 0, stream>>>(x_bf, Wxh0_bf, bxh0, xp0);

  // ---- whole recurrence in ONE persistent kernel, 1 barrier per step ----
  scan_kernel<<<24, 256, 0, stream>>>(xp0,
                                      Whh0_bf, bhh0,
                                      Wxh1_bf, bxh1,
                                      Whh1_bf, bhh1,
                                      h0b0, h0b1, h1b0, h1b1,
                                      h0f32, h1f32, out,
                                      sync + 0, sync + 1);

  // ---- h_final = stack([h0, h1]) appended after outputs ----
  final_kernel<<<(B_ * H_ + 255) / 256, 256, 0, stream>>>(
      h0f32, h1f32, out + (size_t)B_ * S_ * H_);
}
